// SelfAttentionBlock_31756988187040
// MI455X (gfx1250) — matrix operand order repaired and checked by reading, not compiled, not run on hardware
//
#include <hip/hip_runtime.h>
#include <hip/hip_bf16.h>

// Shapes (fixed by reference)
#define BATCH 4
#define CIN   256
#define NTOK  4096   // 16*16*16
#define CT    128
#define COUT  256

typedef _Float16 v16h __attribute__((ext_vector_type(16)));
typedef _Float16 v8h  __attribute__((ext_vector_type(8)));
typedef float    v8f  __attribute__((ext_vector_type(8)));

#define WMMA_F16(a, b, c) \
  __builtin_amdgcn_wmma_f32_16x16x32_f16(false, (a), false, (b), (short)0, (c), false, false)

// ---------------------------------------------------------------------------
// Kernel 0: one-shot f32 -> f16 conversion of the four weight matrices so the
// GEMM kernels do pure vector f16 fragment loads (no per-element cvt in loops).
// Each weight is 128*256 = 32768 elements.
// ---------------------------------------------------------------------------
__global__ __launch_bounds__(256) void cvt_weights_kernel(
    const float* __restrict__ Wq, const float* __restrict__ Wk,
    const float* __restrict__ Wv, const float* __restrict__ Wo,
    _Float16* __restrict__ Wqh, _Float16* __restrict__ Wkh,
    _Float16* __restrict__ Wvh, _Float16* __restrict__ Woh)
{
  const int i = blockIdx.x * 256 + threadIdx.x;   // grid covers 32768
  Wqh[i] = (_Float16)Wq[i];
  Wkh[i] = (_Float16)Wk[i];
  Wvh[i] = (_Float16)Wv[i];
  Woh[i] = (_Float16)Wo[i];
}

// ---------------------------------------------------------------------------
// Kernel 1: QKV projections (1x1x1 conv == channel GEMM), f32 acts -> f16 out.
//   mode 0: Q[b,n,t]  = scale * sum_c Wq[t,c] * qf[b,c,n]  -> (B,N,CT)
//   mode 1: Kn[b,n,t] = sum_c Wk[t,c] * kf[b,c,n]          -> (B,N,CT)
//   mode 2: Vt[b,t,n] = sum_c Wv[t,c] * kf[b,c,n]          -> (B,CT,N)
// Activations staged in LDS as f16 tile[n][c]; each of 8 waves owns one
// 16-wide CT tile; K-loop = 256/32 = 8 WMMAs.  1/sqrt(CT) folded into Q.
// ---------------------------------------------------------------------------
__global__ __launch_bounds__(256) void qkv_proj_kernel(
    const float* __restrict__ qf, const float* __restrict__ kf,
    const _Float16* __restrict__ Wqh, const _Float16* __restrict__ Wkh,
    const _Float16* __restrict__ Wvh,
    _Float16* __restrict__ Q, _Float16* __restrict__ Kn,
    _Float16* __restrict__ Vt)
{
  __shared__ _Float16 tile[16][264];   // 16 tokens x 256 channels (+8 pad)

  const int tid  = threadIdx.x;
  const int b    = blockIdx.y;
  const int mode = blockIdx.z;
  const int n0   = blockIdx.x * 16;

  const float* src = (mode == 0) ? qf : kf;
  for (int i = tid; i < 16 * 256; i += 256) {          // coalesced over n
    const int c = i >> 4, n = i & 15;
    tile[n][c] = (_Float16)src[((size_t)(b * CIN + c)) * NTOK + n0 + n];
  }
  __syncthreads();

  const int wave = tid >> 5, lane = tid & 31;
  const int g = lane >> 4, lm = lane & 15;
  const int t0 = wave * 16;             // CT tile per wave (8*16 = 128)

  if (mode != 2) {
    const _Float16* Wt = (mode == 0) ? Wqh : Wkh;
    _Float16* dst      = (mode == 0) ? Q   : Kn;
    const float oscale = (mode == 0) ? 0.08838834764831845f : 1.0f;  // CT^-0.5
    v8f acc = {};
    #pragma unroll
    for (int ks = 0; ks < 8; ++ks) {
      const int c0 = ks * 32;
      // A (16x32, M=n K=c) from LDS: lane row lm, two contiguous runs of 8
      v16h a;
      #pragma unroll
      for (int i = 0; i < 8; ++i) {
        a[i]     = tile[lm][c0 + g * 8 + i];
        a[8 + i] = tile[lm][c0 + 16 + g * 8 + i];
      }
      // B (32x16, K=c N=t): two contiguous runs of 8, same k order as A
      const _Float16* wtr = Wt + (size_t)(t0 + lm) * CIN + c0;
      v8h blo = *(const v8h*)(wtr + g * 8);
      v8h bhi = *(const v8h*)(wtr + 16 + g * 8);
      v16h bf;
      #pragma unroll
      for (int i = 0; i < 8; ++i) { bf[i] = blo[i]; bf[8 + i] = bhi[i]; }
      acc = WMMA_F16(a, bf, acc);
    }
    #pragma unroll
    for (int r = 0; r < 8; ++r)
      dst[((size_t)(b * NTOK + n0 + r + g * 8)) * CT + t0 + lm] =
          (_Float16)(acc[r] * oscale);
  } else {
    // Vt = Wv * kf : D tile M=t rows, N=n cols
    v8f acc = {};
    #pragma unroll
    for (int ks = 0; ks < 8; ++ks) {
      const int c0 = ks * 32;
      // A (16x32, M=t K=c) from f16 weights: two contiguous 16B loads
      const _Float16* wrow = Wvh + (size_t)(t0 + lm) * CIN + c0;
      v8h lo = *(const v8h*)(wrow + g * 8);
      v8h hi = *(const v8h*)(wrow + 16 + g * 8);
      v16h a;
      #pragma unroll
      for (int i = 0; i < 8; ++i) { a[i] = lo[i]; a[8 + i] = hi[i]; }
      // B (32x16, K=c N=n) from LDS tile
      v16h bf;
      #pragma unroll
      for (int i = 0; i < 16; ++i) bf[i] = tile[lm][c0 + (i >> 3) * 16 + g * 8 + (i & 7)];
      acc = WMMA_F16(a, bf, acc);
    }
    #pragma unroll
    for (int r = 0; r < 8; ++r)
      Vt[((size_t)(b * CT + t0 + r + g * 8)) * NTOK + n0 + lm] = (_Float16)acc[r];
  }
}

// ---------------------------------------------------------------------------
// Kernel 2: attention for one 16-query block. CDNA5's 320KB LDS holds the full
// 16 x 4096 score slab; stored as f16 (128 KB) so phase-C A-fragments are raw
// ds_load_b128 pairs with no conversion VALU in the WMMA loop. Exact softmax.
// ---------------------------------------------------------------------------
__global__ __launch_bounds__(256) void attn_kernel(
    const _Float16* __restrict__ Q, const _Float16* __restrict__ Kn,
    const _Float16* __restrict__ Vt, _Float16* __restrict__ Ctx)
{
  extern __shared__ _Float16 Sh[];     // 16 * 4096 f16 = 128 KB (dynamic)
  __shared__ float red[16][16];
  __shared__ float rowsum[16];
  __shared__ float rowmax[16];

  const int tid  = threadIdx.x;
  const int wave = tid >> 5, lane = tid & 31;
  const int g = lane >> 4, lm = lane & 15;
  const int b  = blockIdx.y;
  const int n0 = blockIdx.x * 16;

  // ---- Phase A: S = (scaled Q) . Kn^T, 16 x 4096, written as f16 ----
  v16h qA[4];                           // Q fragments, 4 k-steps over CT=128
  {
    const _Float16* qrow = Q + ((size_t)(b * NTOK + n0 + lm)) * CT;
    #pragma unroll
    for (int ks = 0; ks < 4; ++ks) {
      v8h lo = *(const v8h*)(qrow + ks * 32 + g * 8);
      v8h hi = *(const v8h*)(qrow + ks * 32 + 16 + g * 8);
      #pragma unroll
      for (int i = 0; i < 8; ++i) { qA[ks][i] = lo[i]; qA[ks][8 + i] = hi[i]; }
    }
  }
  for (int mt = wave; mt < NTOK / 16; mt += 8) {
    v8f acc = {};
    const _Float16* krow = Kn + ((size_t)(b * NTOK + mt * 16 + lm)) * CT;
    #pragma unroll
    for (int ks = 0; ks < 4; ++ks) {
      v8h blo = *(const v8h*)(krow + ks * 32 + g * 8);
      v8h bhi = *(const v8h*)(krow + ks * 32 + 16 + g * 8);
      v16h bf;
      #pragma unroll
      for (int i = 0; i < 8; ++i) { bf[i] = blo[i]; bf[8 + i] = bhi[i]; }
      acc = WMMA_F16(qA[ks], bf, acc);
    }
    #pragma unroll
    for (int r = 0; r < 8; ++r)
      Sh[(r + g * 8) * NTOK + mt * 16 + lm] = (_Float16)acc[r];
  }
  __syncthreads();

  // ---- Phase B: exact softmax over each of 16 rows (16 threads/row) ----
  {
    const int r = tid >> 4, j = tid & 15;
    float mx = -3.0e38f;
    for (int c = j; c < NTOK; c += 16) mx = fmaxf(mx, (float)Sh[r * NTOK + c]);
    red[r][j] = mx;
    __syncthreads();
    if (j == 0) {
      float m2 = red[r][0];
      #pragma unroll
      for (int k = 1; k < 16; ++k) m2 = fmaxf(m2, red[r][k]);
      rowmax[r] = m2;
    }
    __syncthreads();
    const float m = rowmax[r];
    float sum = 0.f;
    for (int c = j; c < NTOK; c += 16) {
      const float e = __expf((float)Sh[r * NTOK + c] - m);
      Sh[r * NTOK + c] = (_Float16)e;
      sum += e;
    }
    red[r][j] = sum;
    __syncthreads();
    if (j == 0) {
      float s2 = 0.f;
      #pragma unroll
      for (int k = 0; k < 16; ++k) s2 += red[r][k];
      rowsum[r] = s2;
    }
    __syncthreads();
  }

  // ---- Phase C: ctx = P . V ; wave w owns CT tile w; normalize at the end ----
  {
    const int ct0 = wave * 16;
    v8f acc = {};
    const _Float16* vrow = Vt + ((size_t)(b * CT + ct0 + lm)) * NTOK;
    for (int kb = 0; kb < NTOK / 32; ++kb) {
      // A fragment: 16 x 32 slice of P straight from LDS (f16, no cvt)
      const _Float16* srow = Sh + lm * NTOK + kb * 32;
      v8h lo = *(const v8h*)(srow + g * 8);
      v8h hi = *(const v8h*)(srow + 16 + g * 8);
      v16h a;
      #pragma unroll
      for (int i = 0; i < 8; ++i) { a[i] = lo[i]; a[8 + i] = hi[i]; }
      v8h blo = *(const v8h*)(vrow + kb * 32 + g * 8);
      v8h bhi = *(const v8h*)(vrow + kb * 32 + 16 + g * 8);
      v16h bf;
      #pragma unroll
      for (int i = 0; i < 8; ++i) { bf[i] = blo[i]; bf[8 + i] = bhi[i]; }
      acc = WMMA_F16(a, bf, acc);
    }
    #pragma unroll
    for (int r = 0; r < 8; ++r) {
      const int row = r + g * 8;
      const float o = acc[r] * (1.0f / rowsum[row]);
      Ctx[((size_t)(b * NTOK + n0 + row)) * CT + ct0 + lm] = (_Float16)o;
    }
  }
}

// ---------------------------------------------------------------------------
// Kernel 3: out[b,o,n] = sum_t Wo[o,t] * ctx[b,n,t].  D tile: o rows x n cols.
// All fragments are contiguous f16 vector loads; stores coalesced over n.
// ---------------------------------------------------------------------------
__global__ __launch_bounds__(256) void out_proj_kernel(
    const _Float16* __restrict__ Woh, const _Float16* __restrict__ Ctx,
    float* __restrict__ out)
{
  const int tid  = threadIdx.x;
  const int wave = tid >> 5, lane = tid & 31;
  const int g = lane >> 4, lm = lane & 15;
  const int b  = blockIdx.y;
  const int n0 = blockIdx.x * 16;

  // B fragments (ctx for this n-tile), shared across both o-tiles of the wave
  v16h bf[4];
  const _Float16* crow = Ctx + ((size_t)(b * NTOK + n0 + lm)) * CT;
  #pragma unroll
  for (int ks = 0; ks < 4; ++ks) {
    v8h blo = *(const v8h*)(crow + ks * 32 + g * 8);
    v8h bhi = *(const v8h*)(crow + ks * 32 + 16 + g * 8);
    #pragma unroll
    for (int i = 0; i < 8; ++i) { bf[ks][i] = blo[i]; bf[ks][8 + i] = bhi[i]; }
  }

  for (int ot = wave; ot < COUT / 16; ot += 8) {
    const _Float16* wrowbase = Woh + (size_t)(ot * 16 + lm) * CT;
    v8f acc = {};
    #pragma unroll
    for (int ks = 0; ks < 4; ++ks) {
      const _Float16* wrow = wrowbase + ks * 32;
      v8h lo = *(const v8h*)(wrow + g * 8);
      v8h hi = *(const v8h*)(wrow + 16 + g * 8);
      v16h a;
      #pragma unroll
      for (int i = 0; i < 8; ++i) { a[i] = lo[i]; a[8 + i] = hi[i]; }
      acc = WMMA_F16(a, bf[ks], acc);
    }
    #pragma unroll
    for (int r = 0; r < 8; ++r)
      out[((size_t)(b * COUT + ot * 16 + r + g * 8)) * NTOK + n0 + lm] = acc[r];
  }
}

extern "C" void kernel_launch(void* const* d_in, const int* in_sizes, int n_in,
                              void* d_out, int out_size, void* d_ws, size_t ws_size,
                              hipStream_t stream) {
  (void)in_sizes; (void)n_in; (void)out_size; (void)ws_size;
  const float* qf = (const float*)d_in[0];
  const float* kf = (const float*)d_in[1];
  const float* Wq = (const float*)d_in[2];
  const float* Wk = (const float*)d_in[3];
  const float* Wv = (const float*)d_in[4];
  const float* Wo = (const float*)d_in[5];
  float* out = (float*)d_out;

  // f16 workspace: Q | Kn | Vt | Ctx (4 MB each) then f16 weights (64 KB each)
  const size_t seg  = (size_t)BATCH * NTOK * CT;   // 2,097,152 halves
  const size_t wseg = (size_t)CT * CIN;            // 32,768 halves
  _Float16* ws  = (_Float16*)d_ws;
  _Float16* Q   = ws;
  _Float16* Kn  = ws + seg;
  _Float16* Vt  = ws + 2 * seg;
  _Float16* Ctx = ws + 3 * seg;
  _Float16* Wqh = ws + 4 * seg;
  _Float16* Wkh = Wqh + wseg;
  _Float16* Wvh = Wkh + wseg;
  _Float16* Woh = Wvh + wseg;

  // 0) weights to f16 (one element per thread, 32768 per weight)
  cvt_weights_kernel<<<dim3(wseg / 256), 256, 0, stream>>>(
      Wq, Wk, Wv, Wo, Wqh, Wkh, Wvh, Woh);

  // 1) QKV projections: grid z selects {Q, K, V}; 1/sqrt(CT) folded into Q
  qkv_proj_kernel<<<dim3(NTOK / 16, BATCH, 3), 256, 0, stream>>>(
      qf, kf, Wqh, Wkh, Wvh, Q, Kn, Vt);

  // 2) attention: 128 KB dynamic LDS f16 score slab per workgroup
  attn_kernel<<<dim3(NTOK / 16, BATCH), 256,
                16 * NTOK * sizeof(_Float16), stream>>>(Q, Kn, Vt, Ctx);

  // 3) output projection
  out_proj_kernel<<<dim3(NTOK / 16, BATCH), 256, 0, stream>>>(Wo ? Woh : Woh, Ctx, out);
}
